// MultiHeadedAttention_24696061952078
// MI455X (gfx1250) — matrix operand order repaired and checked by reading, not compiled, not run on hardware
//
#include <hip/hip_runtime.h>
#include <hip/hip_bf16.h>

// ---------------------------------------------------------------------------
// Multi-head attention for MI455X (gfx1250), bf16 WMMA pipeline:
//   1) convert x -> bf16, transpose-convert Wq/Wk/Wv/Wo -> bf16 [n][k]
//   2) WMMA GEMMs: Q(scaled, bf16 [b,h,s,d]), K(bf16 [b,h,s,d]),
//      V(bf16 TRANSPOSED [b,h,d,s])
//   3) flash-attention (online softmax, DPP16 row reductions),
//      ctx -> bf16 [b,s,dim]
//   4) WMMA GEMM: out = ctx @ Wo + bo   (fp32 output)
// ---------------------------------------------------------------------------

#define DIMN   512
#define HEADS  8
#define DEPTH  64
#define BB     2
#define SEQ    4096
#define MTOT   (BB * SEQ)   // 8192 rows

typedef __attribute__((ext_vector_type(16))) __bf16 v16bf;
typedef __attribute__((ext_vector_type(8)))  __bf16 v8bf;
typedef __attribute__((ext_vector_type(8)))  float  v8f;

// A-matrix 16x32 bf16 fragment (per CDNA5 ISA):
// lane half=0 holds K = {0..7, 16..23}, half=1 holds K = {8..15, 24..31},
// row = lane&15. Elements within the vector pack K ascending in two runs of 8.
__device__ __forceinline__ v16bf load_frag_a(const __bf16* base, int half) {
  v8bf lo = *(const v8bf*)(base + 8 * half);
  v8bf hi = *(const v8bf*)(base + 16 + 8 * half);
  return __builtin_shufflevector(lo, hi, 0,1,2,3,4,5,6,7,8,9,10,11,12,13,14,15);
}

// B-matrix 32x16 bf16 fragment: column n = lane&15, lanes 0-15 cover K={0..7,16..23},
// lanes 16-31 cover K={8..15,24..31}. base points at the column's K-contiguous data.
__device__ __forceinline__ v16bf load_frag_b(const __bf16* base, int half) {
  v8bf lo = *(const v8bf*)(base + 8 * half);
  v8bf hi = *(const v8bf*)(base + 16 + 8 * half);
  return __builtin_shufflevector(lo, hi, 0,1,2,3,4,5,6,7,8,9,10,11,12,13,14,15);
}

#define WMMA_BF16(A, B, C) \
  __builtin_amdgcn_wmma_f32_16x16x32_bf16(false, (A), false, (B), (short)0, (C), false, false)

// ---------------------------------------------------------------------------
// DPP16 butterfly reductions across a 16-lane row (both wave32 halves reduce
// independently -- matches the C-fragment row mapping row = vgpr + 8*(lane>=16)).
// quad_perm(1,0,3,2)=0xB1 (xor1), quad_perm(2,3,0,1)=0x4E (xor2),
// row_half_mirror=0x141, row_mirror=0x140. All lanes active -> no bound issues.
// ---------------------------------------------------------------------------
__device__ __forceinline__ float dpp_max16(float x) {
  int v;
  v = __float_as_int(x);
  x = fmaxf(x, __int_as_float(__builtin_amdgcn_update_dpp(v, v, 0xB1, 0xF, 0xF, false)));
  v = __float_as_int(x);
  x = fmaxf(x, __int_as_float(__builtin_amdgcn_update_dpp(v, v, 0x4E, 0xF, 0xF, false)));
  v = __float_as_int(x);
  x = fmaxf(x, __int_as_float(__builtin_amdgcn_update_dpp(v, v, 0x141, 0xF, 0xF, false)));
  v = __float_as_int(x);
  x = fmaxf(x, __int_as_float(__builtin_amdgcn_update_dpp(v, v, 0x140, 0xF, 0xF, false)));
  return x;
}

__device__ __forceinline__ float dpp_sum16(float x) {
  int v;
  v = __float_as_int(x);
  x = x + __int_as_float(__builtin_amdgcn_update_dpp(v, v, 0xB1, 0xF, 0xF, false));
  v = __float_as_int(x);
  x = x + __int_as_float(__builtin_amdgcn_update_dpp(v, v, 0x4E, 0xF, 0xF, false));
  v = __float_as_int(x);
  x = x + __int_as_float(__builtin_amdgcn_update_dpp(v, v, 0x141, 0xF, 0xF, false));
  v = __float_as_int(x);
  x = x + __int_as_float(__builtin_amdgcn_update_dpp(v, v, 0x140, 0xF, 0xF, false));
  return x;
}

// ---------------------------------------------------------------------------
// conversion kernels
// ---------------------------------------------------------------------------
__global__ void cvt_bf16_kernel(const float* __restrict__ src,
                                __bf16* __restrict__ dst, int n) {
  int i = blockIdx.x * blockDim.x + threadIdx.x;
  if (i < n) dst[i] = (__bf16)src[i];
}

// Wt[n][k] = W[k][n]  (W is [K=512][N=512] row-major, x @ W)
__global__ void transpose_cvt_kernel(const float* __restrict__ W,
                                     __bf16* __restrict__ Wt) {
  int i = blockIdx.x * blockDim.x + threadIdx.x;  // 0..262143
  int nn = i >> 9;
  int kk = i & 511;
  Wt[(size_t)nn * DIMN + kk] = (__bf16)W[(size_t)kk * DIMN + nn];
}

// ---------------------------------------------------------------------------
// WMMA GEMM: out[m][n] over A[M=8192][K=512] (bf16) x Wt[n][k] (bf16) + bias
// MODE 0: Q  -> bf16 [b,h,s,d], scaled by 1/sqrt(DEPTH)
// MODE 1: K  -> bf16 [b,h,s,d]
// MODE 2: V  -> bf16 [b,h,d,s] (transposed)
// MODE 3: O  -> fp32 [m][n] (final output)
// block = 256 threads (8 waves); each wave does a 16x64 tile; block: 128x64.
// ---------------------------------------------------------------------------
template <int MODE>
__global__ __launch_bounds__(256)
void gemm_proj_kernel(const __bf16* __restrict__ A,
                      const __bf16* __restrict__ Wt,
                      const float* __restrict__ bias,
                      void* __restrict__ outp) {
  const int lane = threadIdx.x & 31;
  const int wave = threadIdx.x >> 5;
  const int half = lane >> 4;
  const int lr   = lane & 15;
  const int mbase = blockIdx.x * 128 + wave * 16;
  const int nbase = blockIdx.y * 64;

  const __bf16* arow = A + (size_t)(mbase + lr) * DIMN;

  v8f acc[4] = {};

  for (int kc = 0; kc < DIMN; kc += 32) {
    v16bf af = load_frag_a(arow + kc, half);
#pragma unroll
    for (int t = 0; t < 4; ++t) {
      const __bf16* bptr = Wt + (size_t)(nbase + t * 16 + lr) * DIMN + kc;
      v16bf bfr = load_frag_b(bptr, half);
      acc[t] = WMMA_BF16(af, bfr, acc[t]);
    }
  }

#pragma unroll
  for (int t = 0; t < 4; ++t) {
    const int col = nbase + t * 16 + lr;
    const float bval = bias[col];
#pragma unroll
    for (int j = 0; j < 8; ++j) {
      const int m = mbase + j + 8 * half;   // C layout: row = vgpr + 8*half
      float v = acc[t][j] + bval;
      if (MODE == 0 || MODE == 1) {
        if (MODE == 0) v *= 0.125f;  // 1/sqrt(64) folded into Q
        const int b = m >> 12, s = m & (SEQ - 1);
        const int h = col >> 6, d = col & 63;
        ((__bf16*)outp)[(((size_t)(b * HEADS + h)) * SEQ + s) * DEPTH + d] = (__bf16)v;
      } else if (MODE == 2) {
        const int b = m >> 12, s = m & (SEQ - 1);
        const int h = col >> 6, d = col & 63;
        ((__bf16*)outp)[(((size_t)(b * HEADS + h)) * DEPTH + d) * SEQ + s] = (__bf16)v;
      } else {
        ((float*)outp)[(size_t)m * DIMN + col] = v;
      }
    }
  }
}

// ---------------------------------------------------------------------------
// Flash attention: block = 128 threads (4 waves). Each wave owns TWO 16-row
// q blocks (32 q rows) so one set of streamed K/V fragments feeds 2x the
// WMMAs. Streams 32 keys per iteration; online softmax in registers with
// DPP16 row reductions; P relaid out through per-wave LDS staging.
// ---------------------------------------------------------------------------
__global__ __launch_bounds__(128)
void flash_attn_kernel(const __bf16* __restrict__ Q,
                       const __bf16* __restrict__ K,
                       const __bf16* __restrict__ Vt,
                       __bf16* __restrict__ ctx) {
  __shared__ __align__(16) __bf16 Pst[4][2][16 * 32];

  const int lane = threadIdx.x & 31;
  const int wave = threadIdx.x >> 5;
  const int half = lane >> 4;
  const int lr   = lane & 15;

  const int qt = blockIdx.x;  // 0..31 (128-row q tile per block)
  const int h  = blockIdx.y;
  const int b  = blockIdx.z;
  const size_t bh = (size_t)(b * HEADS + h);

  const __bf16* Qbh = Q  + bh * (size_t)SEQ * DEPTH;
  const __bf16* Kbh = K  + bh * (size_t)SEQ * DEPTH;
  const __bf16* Vbh = Vt + bh * (size_t)DEPTH * SEQ;

  const int qbase = qt * 128 + wave * 32;  // first of this wave's 32 q rows

  // Q fragments: 2 row-blocks x 2 depth-chunks (pre-scaled by 1/8).
  v16bf aq[2][2];
#pragma unroll
  for (int r = 0; r < 2; ++r) {
    const __bf16* qrow = Qbh + (size_t)(qbase + r * 16 + lr) * DEPTH;
    aq[r][0] = load_frag_a(qrow, half);
    aq[r][1] = load_frag_a(qrow + 32, half);
  }

  v8f acc[2][4] = {};
  float mrow[2][8], lrow[2][8];
#pragma unroll
  for (int r = 0; r < 2; ++r)
#pragma unroll
    for (int j = 0; j < 8; ++j) { mrow[r][j] = -1e30f; lrow[r][j] = 0.0f; }

  for (int kt = 0; kt < SEQ; kt += 32) {
    // ---- K fragments for this 32-key tile (shared by both row-blocks) ----
    const __bf16* k0 = Kbh + (size_t)(kt + lr) * DEPTH;        // keys kt..kt+15
    const __bf16* k1 = Kbh + (size_t)(kt + 16 + lr) * DEPTH;   // keys kt+16..kt+31
    const v16bf bk0a = load_frag_b(k0, half);        // d 0..31
    const v16bf bk0b = load_frag_b(k0 + 32, half);   // d 32..63
    const v16bf bk1a = load_frag_b(k1, half);
    const v16bf bk1b = load_frag_b(k1 + 32, half);

#pragma unroll
    for (int r = 0; r < 2; ++r) {
      // ---- logits: S = Q . K^T ----
      v8f s0 = {}, s1 = {};
      s0 = WMMA_BF16(aq[r][0], bk0a, s0);
      s0 = WMMA_BF16(aq[r][1], bk0b, s0);
      s1 = WMMA_BF16(aq[r][0], bk1a, s1);
      s1 = WMMA_BF16(aq[r][1], bk1b, s1);

      // ---- online softmax (row = vgpr j + 8*half; DPP16 reductions) ----
#pragma unroll
      for (int j = 0; j < 8; ++j) {
        const float tmax = dpp_max16(fmaxf(s0[j], s1[j]));
        const float mn   = fmaxf(mrow[r][j], tmax);
        const float fsc  = __expf(mrow[r][j] - mn);
        mrow[r][j] = mn;
        lrow[r][j] *= fsc;
        acc[r][0][j] *= fsc; acc[r][1][j] *= fsc;
        acc[r][2][j] *= fsc; acc[r][3][j] *= fsc;
        const float p0 = __expf(s0[j] - mn);
        const float p1 = __expf(s1[j] - mn);
        s0[j] = p0;
        s1[j] = p1;
        lrow[r][j] += dpp_sum16(p0 + p1);
      }

      // ---- relayout P: C-fragment -> LDS row-major [16 q][32 keys] ----
      __bf16* P = &Pst[wave][r][0];
#pragma unroll
      for (int j = 0; j < 8; ++j) {
        const int rw = j + 8 * half;
        P[rw * 32 + lr]      = (__bf16)s0[j];
        P[rw * 32 + 16 + lr] = (__bf16)s1[j];
      }
    }
    __syncthreads();

    // ---- V fragments for this key tile (shared by both row-blocks) ----
#pragma unroll
    for (int t = 0; t < 4; ++t) {
      const __bf16* vptr = Vbh + (size_t)(t * 16 + lr) * SEQ + kt;  // d = t*16+lr
      const v16bf bv = load_frag_b(vptr, half);
#pragma unroll
      for (int r = 0; r < 2; ++r) {
        const v16bf ap = load_frag_a(&Pst[wave][r][0] + lr * 32, half);
        acc[r][t] = WMMA_BF16(ap, bv, acc[r][t]);
      }
    }
    __syncthreads();  // protect Pst before next iteration's stores
  }

  // ---- normalize + store ctx (merged heads: [b, s, h*64+d] bf16) ----
#pragma unroll
  for (int r = 0; r < 2; ++r) {
#pragma unroll
    for (int t = 0; t < 4; ++t) {
#pragma unroll
      for (int j = 0; j < 8; ++j) {
        const int srow = qbase + r * 16 + j + 8 * half;
        const float rn = 1.0f / lrow[r][j];
        ctx[((size_t)b * SEQ + srow) * DIMN + h * DEPTH + t * 16 + lr] =
            (__bf16)(acc[r][t][j] * rn);
      }
    }
  }
}

// ---------------------------------------------------------------------------
// launch
// ---------------------------------------------------------------------------
extern "C" void kernel_launch(void* const* d_in, const int* in_sizes, int n_in,
                              void* d_out, int out_size, void* d_ws, size_t ws_size,
                              hipStream_t stream) {
  (void)in_sizes; (void)n_in; (void)out_size; (void)ws_size;

  const float* x  = (const float*)d_in[0];
  const float* Wq = (const float*)d_in[1];
  const float* bq = (const float*)d_in[2];
  const float* Wk = (const float*)d_in[3];
  const float* bk = (const float*)d_in[4];
  const float* Wv = (const float*)d_in[5];
  const float* bv = (const float*)d_in[6];
  const float* Wo = (const float*)d_in[7];
  const float* bo = (const float*)d_in[8];

  char* ws = (char*)d_ws;
  __bf16* xbf = (__bf16*)ws; ws += (size_t)MTOT * DIMN * 2;   // 8 MB
  __bf16* Wtq = (__bf16*)ws; ws += (size_t)DIMN * DIMN * 2;   // 512 KB
  __bf16* Wtk = (__bf16*)ws; ws += (size_t)DIMN * DIMN * 2;
  __bf16* Wtv = (__bf16*)ws; ws += (size_t)DIMN * DIMN * 2;
  __bf16* Wto = (__bf16*)ws; ws += (size_t)DIMN * DIMN * 2;
  __bf16* Qb  = (__bf16*)ws; ws += (size_t)MTOT * DIMN * 2;   // 8 MB
  __bf16* Kb  = (__bf16*)ws; ws += (size_t)MTOT * DIMN * 2;
  __bf16* Vtb = (__bf16*)ws; ws += (size_t)MTOT * DIMN * 2;
  __bf16* ctx = (__bf16*)ws; ws += (size_t)MTOT * DIMN * 2;

  // 1) conversions
  {
    const int n = MTOT * DIMN;  // 4,194,304
    cvt_bf16_kernel<<<n / 256, 256, 0, stream>>>(x, xbf, n);
    const int nw = DIMN * DIMN;  // 262,144
    transpose_cvt_kernel<<<nw / 256, 256, 0, stream>>>(Wq, Wtq);
    transpose_cvt_kernel<<<nw / 256, 256, 0, stream>>>(Wk, Wtk);
    transpose_cvt_kernel<<<nw / 256, 256, 0, stream>>>(Wv, Wtv);
    transpose_cvt_kernel<<<nw / 256, 256, 0, stream>>>(Wo, Wto);
  }

  // 2) QKV projections
  dim3 ggrid(MTOT / 128, DIMN / 64);  // 64 x 8
  gemm_proj_kernel<0><<<ggrid, 256, 0, stream>>>(xbf, Wtq, bq, (void*)Qb);
  gemm_proj_kernel<1><<<ggrid, 256, 0, stream>>>(xbf, Wtk, bk, (void*)Kb);
  gemm_proj_kernel<2><<<ggrid, 256, 0, stream>>>(xbf, Wtv, bv, (void*)Vtb);

  // 3) attention (each block covers 128 q rows: 4 waves x 32 rows)
  dim3 agrid(SEQ / 128, HEADS, BB);  // 32 x 8 x 2
  flash_attn_kernel<<<agrid, 128, 0, stream>>>(Qb, Kb, Vtb, ctx);

  // 4) output projection (fp32 out)
  gemm_proj_kernel<3><<<ggrid, 256, 0, stream>>>(ctx, Wto, bo, d_out);
}
